// Feat3dNet_40450001994199
// MI455X (gfx1250) — matrix operand order repaired and checked by reading, not compiled, not run on hardware
//
#include <hip/hip_runtime.h>
#include <hip/hip_bf16.h>

typedef __attribute__((ext_vector_type(16))) _Float16 v16h;
typedef __attribute__((ext_vector_type(8)))  _Float16 v8h;
typedef __attribute__((ext_vector_type(8)))  float    v8f;

#define BATCH 8
#define NPTS  16384
#define NCLUS 512
#define NSAMP 64
#define R2    4.0f
#define RINV  0.5f

// ---------------------------------------------------------------------------
// Kernel 1: farthest point sampling. One block per batch; 1024 threads own
// 16 points each (coords + running min-dist in registers). 512 sequential
// argmax steps via LDS tree reduction. Also writes new_xyz output.
// ---------------------------------------------------------------------------
__global__ __launch_bounds__(1024)
void fps_kernel(const float* __restrict__ xyz, float* __restrict__ out_newxyz) {
    const int b   = blockIdx.x;
    const int tid = threadIdx.x;
    const float* base = xyz + (size_t)b * NPTS * 3;

    float px[16], py[16], pz[16], dist[16];
#pragma unroll
    for (int t = 0; t < 16; ++t) {
        int p = tid + t * 1024;
        px[t] = base[p * 3 + 0];
        py[t] = base[p * 3 + 1];
        pz[t] = base[p * 3 + 2];
        dist[t] = 3.4e38f;
    }

    __shared__ float rv[1024];
    __shared__ int   ri[1024];
    __shared__ int   curSh;

    int cur = 0;
    for (int s = 0; s < NCLUS; ++s) {
        float lx = base[cur * 3 + 0];
        float ly = base[cur * 3 + 1];
        float lz = base[cur * 3 + 2];
        if (tid == 0) {
            float* o = out_newxyz + ((size_t)b * NCLUS + s) * 3;
            o[0] = lx; o[1] = ly; o[2] = lz;
        }
        float bv = -1.0f; int bi = 0;
#pragma unroll
        for (int t = 0; t < 16; ++t) {
            float dx = px[t] - lx, dy = py[t] - ly, dz = pz[t] - lz;
            float d  = dx * dx + dy * dy + dz * dz;
            dist[t]  = fminf(dist[t], d);
            if (dist[t] > bv) { bv = dist[t]; bi = tid + t * 1024; }
        }
        rv[tid] = bv; ri[tid] = bi;
        __syncthreads();
        for (int off = 512; off > 0; off >>= 1) {
            if (tid < off) {
                float v2 = rv[tid + off]; int i2 = ri[tid + off];
                if (v2 > rv[tid] || (v2 == rv[tid] && i2 < ri[tid])) {
                    rv[tid] = v2; ri[tid] = i2;
                }
            }
            __syncthreads();
        }
        if (tid == 0) curSh = ri[0];
        __syncthreads();
        cur = curSh;
        __syncthreads();
    }
}

// ---------------------------------------------------------------------------
// Kernel 2: ball query. One wave per cluster center; ordered scan in chunks
// of 32 via ballot + prefix popcount -> ascending indices, matching the
// reference's sort. Pads with first hit (or N-1 if none).
// ---------------------------------------------------------------------------
__global__ __launch_bounds__(128)
void ballq_kernel(const float* __restrict__ xyz,
                  const float* __restrict__ newxyz,
                  int* __restrict__ idxw, float* __restrict__ idxf) {
    const int wave = threadIdx.x >> 5;
    const int lane = threadIdx.x & 31;
    const int g    = blockIdx.x * 4 + wave;      // 0..4095
    const int b    = g >> 9;

    const float cx = newxyz[g * 3 + 0];
    const float cy = newxyz[g * 3 + 1];
    const float cz = newxyz[g * 3 + 2];
    const float* base = xyz + (size_t)b * NPTS * 3;

    int cnt = 0, first = -1;
    for (int s0 = 0; s0 < NPTS && cnt < NSAMP; s0 += 32) {
        int i = s0 + lane;
        float dx = base[i * 3 + 0] - cx;
        float dy = base[i * 3 + 1] - cy;
        float dz = base[i * 3 + 2] - cz;
        bool pred = (dx * dx + dy * dy + dz * dz) < R2;
        unsigned m = (unsigned)__ballot(pred);
        if (pred) {
            int r = cnt + __popc(m & ((1u << lane) - 1u));
            if (r < NSAMP) {
                idxw[(size_t)g * NSAMP + r] = i;
                idxf[(size_t)g * NSAMP + r] = (float)i;
            }
        }
        if (first < 0 && m) first = s0 + __builtin_ctz(m);
        cnt += __popc(m);
    }
    if (cnt > NSAMP) cnt = NSAMP;
    int pad = (cnt > 0) ? first : (NPTS - 1);
    for (int r = cnt + lane; r < NSAMP; r += 32) {
        idxw[(size_t)g * NSAMP + r] = pad;
        idxf[(size_t)g * NSAMP + r] = (float)pad;
    }
}

// ---------------------------------------------------------------------------
// Weight pre-pack: f32 [K][N] -> f16 fragment-major, zero-padded past Kdim.
// Element t of lane L in block (nt,kc) holds w[(kc*32 + (t&7) + 8*(L>=16) +
// 16*(t>=8)) * N + nt*16 + (L&15)], i.e. exactly the B-matrix 32x16 wave32 WMMA layout, so
// the GEMM loads one lane's 16 halfs as two contiguous b128 loads.
// ---------------------------------------------------------------------------
__global__ __launch_bounds__(256)
void packw_kernel(const float* __restrict__ w, _Float16* __restrict__ wp,
                  int Kdim, int kch, int N) {
    int e = blockIdx.x * 256 + threadIdx.x;
    int total = (N >> 4) * kch * 512;
    if (e >= total) return;
    int t    = e & 15;
    int lane = (e >> 4) & 31;
    int rest = e >> 9;
    int kc   = rest % kch;
    int nt   = rest / kch;
    int K    = kc * 32 + (t & 7) + ((lane & 16) ? 8 : 0) + ((t & 8) ? 16 : 0);
    int col  = nt * 16 + (lane & 15);
    wp[e] = (K < Kdim) ? (_Float16)w[K * N + col] : (_Float16)0.f;
}

// ---------------------------------------------------------------------------
// WMMA GEMM layer, compile-time shapes (M=64 fixed, 4 waves = 4 row bands).
// Wave `w` owns row tile mt=w; its A fragments (KCH x v16h) are loaded from
// LDS ONCE and reused across all N tiles, so the inner loop is just
// 2 global b128 loads + 1 WMMA per K-chunk. The nt loop has a compile-time
// trip count -> no EXEC-masked loop around the WMMA (wave-uniform control).
// Wave32 layouts per CDNA5 ISA 7.12.2.
// ---------------------------------------------------------------------------
template <int KD, int ND>
__device__ __forceinline__
void gemm_layer(const _Float16* __restrict__ in,   // [64][KD] LDS, 16B aligned
                _Float16* __restrict__ outp,       // [64][ND] LDS
                const _Float16* __restrict__ wp,   // packed weights
                const float* __restrict__ bias,
                int lane, int wave) {
    constexpr int KCH = KD >> 5;
    constexpr int NT  = ND >> 4;
    const int rsel = (lane & 16) ? 8 : 0;
    const int mt   = wave;                         // fixed row band per wave
    const int arow = mt * 16 + (lane & 15);
    const int ncol = lane & 15;

    v16h afrag[KCH];
#pragma unroll
    for (int kc = 0; kc < KCH; ++kc) {
        const _Float16* ap = in + arow * KD + kc * 32 + rsel;
        v8h alo = *(const v8h*)(ap);
        v8h ahi = *(const v8h*)(ap + 16);
        afrag[kc] = __builtin_shufflevector(alo, ahi,
            0, 1, 2, 3, 4, 5, 6, 7, 8, 9, 10, 11, 12, 13, 14, 15);
    }

    for (int nt = 0; nt < NT; ++nt) {
        const int col = nt * 16 + ncol;
        v8f acc = {};
#pragma unroll
        for (int kc = 0; kc < KCH; ++kc) {
            const _Float16* bp = wp + (((nt * KCH) + kc) * 32 + lane) * 16;
            v8h blo = *(const v8h*)(bp);
            v8h bhi = *(const v8h*)(bp + 8);
            v16h bf = __builtin_shufflevector(blo, bhi,
                0, 1, 2, 3, 4, 5, 6, 7, 8, 9, 10, 11, 12, 13, 14, 15);
            acc = __builtin_amdgcn_wmma_f32_16x16x32_f16(
                false, afrag[kc], false, bf, (short)0, acc, false, false);
        }
        const float bb = bias[col];
#pragma unroll
        for (int r = 0; r < 8; ++r) {
            int row = mt * 16 + r + rsel;
            float v = fmaxf(acc[r] + bb, 0.f);
            outp[row * ND + col] = (_Float16)v;
        }
    }
}

// ---------------------------------------------------------------------------
// Kernel 3: per-cluster MLP. One 128-thread (4-wave) block per cluster.
// Gather + normalize points into zero-padded 64x32 LDS f16 (K=3 -> 32),
// three WMMA GEMM layers ping-ponging LDS, row-max, VALU tail.
// ---------------------------------------------------------------------------
__global__ __launch_bounds__(128)
void mlp_kernel(const float* __restrict__ xyz,
                const float* __restrict__ newxyz,
                const int*   __restrict__ idxw,
                const _Float16* __restrict__ wp0, const float* __restrict__ b0,
                const _Float16* __restrict__ wp1, const float* __restrict__ b1,
                const _Float16* __restrict__ wp2, const float* __restrict__ b2,
                const float* __restrict__ w3, const float* __restrict__ b3,
                const float* __restrict__ w4, const float* __restrict__ b4,
                const float* __restrict__ attw, const float* __restrict__ attb,
                const float* __restrict__ oriw, const float* __restrict__ orib,
                float* __restrict__ out_att, float* __restrict__ out_ori) {
    const int g    = blockIdx.x;             // 0..4095
    const int b    = g >> 9;
    const int tid  = threadIdx.x;
    const int lane = tid & 31;
    const int wave = tid >> 5;

    __shared__ alignas(16) _Float16 sPts[NSAMP * 32]; // [64][32], zero-padded
    __shared__ alignas(16) _Float16 sA[64 * 128];     // ping
    __shared__ alignas(16) _Float16 sB[64 * 256];     // pong
    __shared__ float sM[256];
    __shared__ float sH3[128];
    __shared__ float sH4[64];

    // gather + normalize; each of 64 threads fills one full 32-half row
    if (tid < NSAMP) {
        int i = idxw[(size_t)g * NSAMP + tid];
        const float* p = xyz + ((size_t)b * NPTS + i) * 3;
        const float* c = newxyz + (size_t)g * 3;
        v8h r0 = {};
        r0[0] = (_Float16)((p[0] - c[0]) * RINV);
        r0[1] = (_Float16)((p[1] - c[1]) * RINV);
        r0[2] = (_Float16)((p[2] - c[2]) * RINV);
        v8h z = {};
        v8h* row = (v8h*)(sPts + tid * 32);
        row[0] = r0; row[1] = z; row[2] = z; row[3] = z;
    }
    __syncthreads();

    gemm_layer<32, 64>(sPts, sB, wp0, b0, lane, wave);   // 64x32  -> 64x64
    __syncthreads();
    gemm_layer<64, 128>(sB, sA, wp1, b1, lane, wave);    // 64x64  -> 64x128
    __syncthreads();
    gemm_layer<128, 256>(sA, sB, wp2, b2, lane, wave);   // 64x128 -> 64x256
    __syncthreads();

    // max over the 64 samples
    for (int c = tid; c < 256; c += 128) {
        float m = -3.4e38f;
        for (int r = 0; r < NSAMP; ++r)
            m = fmaxf(m, (float)sB[r * 256 + c]);
        sM[c] = m;
    }
    __syncthreads();

    // h3 = relu(m . w3 + b3), 256 -> 128
    if (tid < 128) {
        float acc = b3[tid];
        for (int c = 0; c < 256; ++c) acc += sM[c] * w3[c * 128 + tid];
        sH3[tid] = fmaxf(acc, 0.f);
    }
    __syncthreads();

    // h4 = relu(h3 . w4 + b4), 128 -> 64
    if (tid < 64) {
        float acc = b4[tid];
        for (int d = 0; d < 128; ++d) acc += sH3[d] * w4[d * 64 + tid];
        sH4[tid] = fmaxf(acc, 0.f);
    }
    __syncthreads();

    // heads
    if (tid == 0) {
        float att = attb[0], ox = orib[0], oy = orib[1];
        for (int e = 0; e < 64; ++e) {
            float h = sH4[e];
            att += h * attw[e];
            ox  += h * oriw[e * 2 + 0];
            oy  += h * oriw[e * 2 + 1];
        }
        float sp = fmaxf(att, 0.f) + log1pf(expf(-fabsf(att)));
        out_att[g] = sp;
        float inv = rsqrtf(fmaxf(ox * ox + oy * oy, 1e-8f));
        out_ori[g] = atan2f(oy * inv, ox * inv);
    }
}

// ---------------------------------------------------------------------------
// Host launcher. Output layout (floats, tuple order):
//   [0      , 12288 )  new_xyz  [8,512,3]
//   [12288  , 274432)  idx      [8,512,64] (index values as float)
//   [274432 , 278528)  attention[8,512]
//   [278528 , 282624)  orientation[8,512]
// Workspace: int idx scratch [8*512*64] (1 MiB) + packed f16 weights.
// ---------------------------------------------------------------------------
extern "C" void kernel_launch(void* const* d_in, const int* in_sizes, int n_in,
                              void* d_out, int out_size, void* d_ws, size_t ws_size,
                              hipStream_t stream) {
    const float* xyz  = (const float*)d_in[0];
    const float* w0   = (const float*)d_in[1];
    const float* b0   = (const float*)d_in[2];
    const float* w1   = (const float*)d_in[3];
    const float* b1   = (const float*)d_in[4];
    const float* w2   = (const float*)d_in[5];
    const float* b2   = (const float*)d_in[6];
    const float* w3   = (const float*)d_in[7];
    const float* b3   = (const float*)d_in[8];
    const float* w4   = (const float*)d_in[9];
    const float* b4   = (const float*)d_in[10];
    const float* attw = (const float*)d_in[11];
    const float* attb = (const float*)d_in[12];
    const float* oriw = (const float*)d_in[13];
    const float* orib = (const float*)d_in[14];

    float* out      = (float*)d_out;
    float* out_nxyz = out;
    float* out_idxf = out + 12288;
    float* out_att  = out + 274432;
    float* out_ori  = out + 278528;

    int*       idxw = (int*)d_ws;
    _Float16*  wp0  = (_Float16*)((char*)d_ws + (1u << 20));
    _Float16*  wp1  = wp0 + 4 * 1 * 512;   // 2048
    _Float16*  wp2  = wp1 + 8 * 2 * 512;   // 8192

    // pack weights to WMMA-B fragment-major f16 (tiny; L2-resident afterwards)
    packw_kernel<<<(4 * 1 * 512 + 255) / 256, 256, 0, stream>>>(w0, wp0, 3, 1, 64);
    packw_kernel<<<(8 * 2 * 512 + 255) / 256, 256, 0, stream>>>(w1, wp1, 64, 2, 128);
    packw_kernel<<<(16 * 4 * 512 + 255) / 256, 256, 0, stream>>>(w2, wp2, 128, 4, 256);

    fps_kernel<<<BATCH, 1024, 0, stream>>>(xyz, out_nxyz);
    ballq_kernel<<<(BATCH * NCLUS) / 4, 128, 0, stream>>>(xyz, out_nxyz, idxw, out_idxf);
    mlp_kernel<<<BATCH * NCLUS, 128, 0, stream>>>(
        xyz, out_nxyz, idxw,
        wp0, b0, wp1, b1, wp2, b2, w3, b3, w4, b4,
        attw, attb, oriw, orib, out_att, out_ori);
}
